// LinearMoE_60816736911603
// MI455X (gfx1250) — matrix operand order repaired and checked by reading, not compiled, not run on hardware
//
#include <hip/hip_runtime.h>
#include <hip/hip_bf16.h>

// ---------------- types ----------------
typedef __attribute__((ext_vector_type(16))) __bf16    v16bf;
typedef __attribute__((ext_vector_type(8)))  float     v8f;
typedef __attribute__((ext_vector_type(4)))  unsigned  u32x4;
typedef __attribute__((ext_vector_type(4)))  float     f32x4;
typedef __attribute__((ext_vector_type(4)))  int       v4i;

#define D_IN   1024
#define D_OUT  1024
#define NEXP   8
#define RANK   32
#define ER     (NEXP * RANK)        // 256
#define T_TOT  65536                // 4 * 16384 tokens
#define LDSTR  40                   // padded LDS row stride in halves (80B, 16B-aligned, conflict-free)
#define KSTEPS (D_IN / 32)          // 32 K-iterations

// ---------------- async copy (memory -> LDS, no VGPR round trip) ----------------
#if __has_builtin(__builtin_amdgcn_global_load_async_to_lds_b128)
#define ASYNC_CP16(g, l)                                                        \
  __builtin_amdgcn_global_load_async_to_lds_b128(                               \
      (__attribute__((address_space(1))) v4i*)(g),                              \
      (__attribute__((address_space(3))) v4i*)(l), 0, 0)
#if __has_builtin(__builtin_amdgcn_s_wait_asynccnt)
#define ASYNC_WAIT() __builtin_amdgcn_s_wait_asynccnt(0)
#else
#define ASYNC_WAIT() asm volatile("s_wait_asynccnt 0" ::: "memory")
#endif
#else
// fallback: synchronous copy through VGPRs
#define ASYNC_CP16(g, l) (*(u32x4*)(l) = *(const u32x4*)(g))
#define ASYNC_WAIT() ((void)0)
#endif

// ---------------- helpers ----------------
__device__ __forceinline__ unsigned short f32_to_bf16_rne(float f) {
  union { float f; unsigned u; } c; c.f = f;
  unsigned u = c.u;
  u += 0x7fffu + ((u >> 16) & 1u);
  return (unsigned short)(u >> 16);
}

// issue async copies for a 64-row x 32-half bf16 tile (256 threads, one b128 each)
__device__ __forceinline__ void async_tile64(const unsigned short* __restrict__ g,
                                             int ld, unsigned short* __restrict__ lds) {
  const int row = threadIdx.x >> 2;
  const int c8  = (threadIdx.x & 3) << 3;           // half offset, 16B chunks
  ASYNC_CP16(g + (long long)row * ld + c8, lds + row * LDSTR + c8);
}

// issue async copies for a 256-row x 32-half bf16 tile (4 b128 per thread)
__device__ __forceinline__ void async_tile256(const unsigned short* __restrict__ g,
                                              int ld, unsigned short* __restrict__ lds) {
  const int c8 = (threadIdx.x & 3) << 3;
#pragma unroll
  for (int j = 0; j < 4; ++j) {
    const int row = (threadIdx.x >> 2) + 64 * j;
    ASYNC_CP16(g + (long long)row * ld + c8, lds + row * LDSTR + c8);
  }
}

// A fragment: 16x32 bf16. lane L: row M = L&15; halves 0..7 = K k0..k0+7,
// halves 8..15 = K 16+k0..16+k0+7 with k0 = (L>=16)?8:0  (ISA 7.12.2)
__device__ __forceinline__ v16bf load_frag_a(const unsigned short* __restrict__ lds,
                                             int rowBase) {
  const int lane = threadIdx.x & 31;
  const int m  = lane & 15;
  const int k0 = (lane >> 4) << 3;
  const unsigned short* p = lds + (rowBase + m) * LDSTR;
  union { v16bf v; u32x4 q[2]; } u;
  u.q[0] = *(const u32x4*)(p + k0);
  u.q[1] = *(const u32x4*)(p + 16 + k0);
  return u.v;
}

// B fragment: 32x16 bf16 (KxN). lane L: col N = L&15; halves 0..7 = K k0..k0+7,
// halves 8..15 = K 16+k0..16+k0+7, k0 = (L>=16)?8:0. B[k][n] = Wrow[n][k] -> contiguous row in LDS tile.
__device__ __forceinline__ v16bf load_frag_b(const unsigned short* __restrict__ lds,
                                             int colBase) {
  const int lane = threadIdx.x & 31;
  const int n  = lane & 15;
  const int k0 = (lane >> 4) << 3;
  const unsigned short* p = lds + (colBase + n) * LDSTR + k0;
  union { v16bf v; u32x4 q[2]; } u;
  u.q[0] = *(const u32x4*)(p);
  u.q[1] = *(const u32x4*)(p + 16);
  return u.v;
}

__device__ __forceinline__ v8f wmma_bf16(v16bf a, v16bf b, v8f c) {
  return __builtin_amdgcn_wmma_f32_16x16x32_bf16(false, a, false, b, (short)0, c,
                                                 false, false);
}

// ---------------- kernel 0: one-pass f32 -> bf16 conversion ----------------
// 8 elements per thread; count must be a multiple of 2048.
__global__ __launch_bounds__(256)
void cvt_bf16_kernel(const float* __restrict__ src, unsigned short* __restrict__ dst) {
  const long long i = ((long long)blockIdx.x * 256 + threadIdx.x) * 8;
  const f32x4 f0 = *(const f32x4*)(src + i);
  const f32x4 f1 = *(const f32x4*)(src + i + 4);
  union { unsigned short h[8]; u32x4 q; } u;
  u.h[0] = f32_to_bf16_rne(f0.x); u.h[1] = f32_to_bf16_rne(f0.y);
  u.h[2] = f32_to_bf16_rne(f0.z); u.h[3] = f32_to_bf16_rne(f0.w);
  u.h[4] = f32_to_bf16_rne(f1.x); u.h[5] = f32_to_bf16_rne(f1.y);
  u.h[6] = f32_to_bf16_rne(f1.z); u.h[7] = f32_to_bf16_rne(f1.w);
  *(u32x4*)(dst + i) = u.q;
}

// ---------------- kernel 1: H[T,256] = x @ A_all^T (bf16 WMMA, async-LDS) ----
// grid.x = T/64; block = 256 (8 waves). Wave w: row-tile r = w&3,
// col-tile group cg = (w>>2)*8 (8 accumulators -> 64x256 tile per block).
__global__ __launch_bounds__(256)
void lora_h_kernel(const unsigned short* __restrict__ xb,
                   const unsigned short* __restrict__ Ab,
                   float* __restrict__ H) {
  __shared__ alignas(16) unsigned short xs[2][64 * LDSTR];
  __shared__ alignas(16) unsigned short as[2][ER * LDSTR];

  const long long t0 = (long long)blockIdx.x * 64;
  const int wave = threadIdx.x >> 5;
  const int r  = wave & 3;
  const int cg = (wave >> 2) * 8;

  v8f acc[8];
  const v8f z = {};
#pragma unroll
  for (int i = 0; i < 8; ++i) acc[i] = z;

  // prologue: stage tile 0
  async_tile64 (xb + t0 * D_IN, D_IN, xs[0]);
  async_tile256(Ab,             D_IN, as[0]);

  for (int it = 0; it < KSTEPS; ++it) {
    const int cur = it & 1;
    ASYNC_WAIT();          // tile 'it' resident in LDS (this wave's copies done)
    __syncthreads();       // all waves' copies visible; prev reads complete
    if (it + 1 < KSTEPS) { // prefetch tile it+1 into the other buffer
      const int kb = (it + 1) * 32;
      async_tile64 (xb + t0 * D_IN + kb, D_IN, xs[cur ^ 1]);
      async_tile256(Ab + kb,             D_IN, as[cur ^ 1]);
    }
    const v16bf af = load_frag_a(xs[cur], r * 16);
#pragma unroll
    for (int c = 0; c < 8; ++c) {
      const v16bf bf = load_frag_b(as[cur], (cg + c) * 16);
      acc[c] = wmma_bf16(af, bf, acc[c]);
    }
  }

  const int lane  = threadIdx.x & 31;
  const int n     = lane & 15;
  const int mBase = (lane >> 4) * 8;
#pragma unroll
  for (int c = 0; c < 8; ++c) {
    const int col = (cg + c) * 16 + n;
#pragma unroll
    for (int i = 0; i < 8; ++i) {
      const int m = r * 16 + mBase + i;
      H[(t0 + m) * ER + col] = acc[c][i];
    }
  }
}

// ---------------- kernel 2: out = x @ W^T + b + LoRA combine ----------------
// grid = (D_OUT/64, T/64); block = 256 (8 waves). Wave w: row-tile r = w&3,
// col-tiles c0, c0+1 with c0 = (w>>2)*2 (2 accumulators -> 64x64 tile).
__global__ __launch_bounds__(256)
void moe_main_kernel(const unsigned short* __restrict__ xb,
                     const unsigned short* __restrict__ Wb,
                     const float* __restrict__ bias,
                     const float* __restrict__ loraB,
                     const int* __restrict__ idx, const float* __restrict__ H,
                     float* __restrict__ out) {
  __shared__ alignas(16) unsigned short xs[2][64 * LDSTR];
  __shared__ alignas(16) unsigned short wsl[2][64 * LDSTR];
  __shared__ alignas(16) float Hs[64][2 * RANK];   // active-expert H slices
  __shared__ int eids[64][2];

  const long long t0 = (long long)blockIdx.y * 64;
  const int o0 = blockIdx.x * 64;
  const int wave = threadIdx.x >> 5;
  const int r  = wave & 3;
  const int c0 = (wave >> 2) * 2;

  // prologue: stage GEMM tile 0
  async_tile64(xb + t0 * D_IN,            D_IN, xs[0]);
  async_tile64(Wb + (long long)o0 * D_IN, D_IN, wsl[0]);

  // ---- stage per-token expert ids (dedup -> mask 'any' semantics) ----
  if (threadIdx.x < 64) {
    const long long t = t0 + threadIdx.x;
    const int e0 = idx[t * 2 + 0];
    const int e1 = idx[t * 2 + 1];
    eids[threadIdx.x][0] = e0;
    eids[threadIdx.x][1] = (e1 == e0) ? -1 : e1;
  }
  __syncthreads();
  // ---- gather active H slices into LDS (zeros for dup slot) ----
  {
    const int t  = threadIdx.x >> 2;
    const int j0 = (threadIdx.x & 3) * 16;
#pragma unroll
    for (int j = j0; j < j0 + 16; ++j) {
      const int s = j >> 5, rr = j & 31;
      const int e = eids[t][s];
      Hs[t][j] = (e < 0) ? 0.f : H[(t0 + t) * ER + e * RANK + rr];
    }
  }

  v8f acc0 = {}, acc1 = {};
  for (int it = 0; it < KSTEPS; ++it) {
    const int cur = it & 1;
    ASYNC_WAIT();
    __syncthreads();
    if (it + 1 < KSTEPS) {
      const int kb = (it + 1) * 32;
      async_tile64(xb + t0 * D_IN + kb,            D_IN, xs[cur ^ 1]);
      async_tile64(Wb + (long long)o0 * D_IN + kb, D_IN, wsl[cur ^ 1]);
    }
    const v16bf af  = load_frag_a(xs[cur], r * 16);
    const v16bf bf0 = load_frag_b(wsl[cur], (c0 + 0) * 16);
    const v16bf bf1 = load_frag_b(wsl[cur], (c0 + 1) * 16);
    acc0 = wmma_bf16(af, bf0, acc0);
    acc1 = wmma_bf16(af, bf1, acc1);
  }

  // ---- epilogue: bias + masked LoRA combine, single store of out ----
  const int lane  = threadIdx.x & 31;
  const int n     = lane & 15;
  const int mBase = (lane >> 4) * 8;
#pragma unroll
  for (int tc = 0; tc < 2; ++tc) {
    const v8f acc = tc ? acc1 : acc0;
    const int o = o0 + (c0 + tc) * 16 + n;
    const float bo = bias[o];
#pragma unroll
    for (int i = 0; i < 8; ++i) {
      const int mloc = r * 16 + mBase + i;
      float v = acc[i] + bo;
#pragma unroll
      for (int s = 0; s < 2; ++s) {
        const int e  = eids[mloc][s];
        const int es = (e < 0) ? 0 : e;           // Hs slot is zero when e<0
        const float* Bp = loraB + ((long long)es * D_OUT + o) * RANK;
        const float* hp = &Hs[mloc][s * RANK];
        float sum = 0.f;
#pragma unroll
        for (int rr = 0; rr < RANK; rr += 4) {
          const f32x4 bb = *(const f32x4*)(Bp + rr);
          sum += hp[rr] * bb.x + hp[rr + 1] * bb.y +
                 hp[rr + 2] * bb.z + hp[rr + 3] * bb.w;
        }
        v += sum;
      }
      out[(t0 + mloc) * (long long)D_OUT + o] = v;
    }
  }
}

// ---------------- launch ----------------
extern "C" void kernel_launch(void* const* d_in, const int* in_sizes, int n_in,
                              void* d_out, int out_size, void* d_ws, size_t ws_size,
                              hipStream_t stream) {
  const float* x  = (const float*)d_in[0];   // [4,16384,1024]
  const float* W  = (const float*)d_in[1];   // [1024,1024]
  const float* b  = (const float*)d_in[2];   // [1024]
  const float* lA = (const float*)d_in[3];   // [8,32,1024] == [256,1024]
  const float* lB = (const float*)d_in[4];   // [8,1024,32]
  const int*  idx = (const int*)d_in[5];     // [4,16384,2]
  float* out = (float*)d_out;                // [65536,1024]

  // workspace layout (bytes): xb 128M | Wb 2M | Ab 0.5M | H 64M
  unsigned short* xb = (unsigned short*)d_ws;
  unsigned short* Wb = xb + (size_t)T_TOT * D_IN;
  unsigned short* Ab = Wb + (size_t)D_OUT * D_IN;
  float*          H  = (float*)(Ab + (size_t)ER * D_IN);

  cvt_bf16_kernel<<<dim3((T_TOT * D_IN) / 2048), 256, 0, stream>>>(x,  xb);
  cvt_bf16_kernel<<<dim3((D_OUT * D_IN) / 2048), 256, 0, stream>>>(W,  Wb);
  cvt_bf16_kernel<<<dim3((ER * D_IN)    / 2048), 256, 0, stream>>>(lA, Ab);

  lora_h_kernel<<<dim3(T_TOT / 64), 256, 0, stream>>>(xb, Ab, H);
  moe_main_kernel<<<dim3(D_OUT / 64, T_TOT / 64), 256, 0, stream>>>(
      xb, Wb, b, lB, idx, H, out);
}
